// GroupedQueryAttention_64278480552138
// MI455X (gfx1250) — matrix operand order repaired and checked by reading, not compiled, not run on hardware
//
#include <hip/hip_runtime.h>
#include <hip/hip_bf16.h>

typedef unsigned short u16;
typedef __attribute__((ext_vector_type(16))) __bf16 v16bf;
typedef __attribute__((ext_vector_type(8)))  float  v8f;

struct __attribute__((aligned(16))) U128 { unsigned int x, y, z, w; };
union FragU { U128 h[2]; v16bf v; };

#define S_LEN 2048
#define DMODEL 2048
#define DKV 512
#define HD 64
#define NHEADS 32
#define NGROUPS 8

static __device__ __forceinline__ u16 f32_to_bf16(float f) {
    unsigned int u = __float_as_uint(f);
    unsigned int r = (u + 0x7FFFu + ((u >> 16) & 1u)) >> 16;
    return (u16)r;
}

static __device__ __forceinline__ v8f splat8(float x) {
    v8f r;
#pragma unroll
    for (int i = 0; i < 8; i++) r[i] = x;
    return r;
}

static __device__ __forceinline__ v8f vmax8(v8f a, v8f b) {
    v8f r;
#pragma unroll
    for (int i = 0; i < 8; i++) r[i] = fmaxf(a[i], b[i]);
    return r;
}

static __device__ __forceinline__ v8f wmma_bf16(v16bf a, v16bf b, v8f c) {
    return __builtin_amdgcn_wmma_f32_16x16x32_bf16(
        /*neg_a=*/false, a, /*neg_b=*/false, b,
        /*c_mod=*/(short)0, c, /*reuse_a=*/false, /*reuse_b=*/false);
}

// CDNA5 async global->LDS copy (16B per lane), tracked by ASYNCcnt.
static __device__ __forceinline__ void async_copy_b128(const void* gsrc, unsigned lds_off) {
    asm volatile("global_load_async_to_lds_b128 %0, %1, off"
                 :: "v"(lds_off), "v"((unsigned long long)(uintptr_t)gsrc)
                 : "memory");
}
static __device__ __forceinline__ void wait_asynccnt0() {
    asm volatile("s_wait_asynccnt 0x0" ::: "memory");
}
// LDS byte offset of a __shared__ object (flat shared addr low 32 bits == LDS offset).
template <typename T>
static __device__ __forceinline__ unsigned lds_off_of(T* p) {
    return (unsigned)(uintptr_t)(void*)p;
}

// ---------------------------------------------------------------------------
// f32 -> bf16 conversion, 4 elements/thread, grid-stride
// ---------------------------------------------------------------------------
__global__ void cvt_bf16_kernel(const float* __restrict__ in, u16* __restrict__ out, int n) {
    int i4 = (blockIdx.x * blockDim.x + threadIdx.x) * 4;
    int stride = gridDim.x * blockDim.x * 4;
    for (; i4 < n; i4 += stride) {
        float4 f = *(const float4*)(in + i4);
        u16 r[4] = { f32_to_bf16(f.x), f32_to_bf16(f.y), f32_to_bf16(f.z), f32_to_bf16(f.w) };
        *(uint2*)(out + i4) = *(uint2*)r;
    }
}

// ---------------------------------------------------------------------------
// bf16 GEMM: C[M,N] = A[M,K] * B[K,N], A/B row-major bf16, C bf16 or f32.
// Block tile 128x256, 8 waves as 2(M) x 4(N); wave tile 64x64 (4x4 WMMA).
// K-step 32. A staged via async global->LDS; B staged transposed manually.
// ---------------------------------------------------------------------------
template <typename OutT>
static __device__ __forceinline__ void store_out(OutT* p, float v);
template <>
__device__ __forceinline__ void store_out<float>(float* p, float v) { *p = v; }
template <>
__device__ __forceinline__ void store_out<u16>(u16* p, float v) { *p = f32_to_bf16(v); }

template <typename OutT>
__global__ void __launch_bounds__(256) gemm_bf16_kernel(
    const u16* __restrict__ A, const u16* __restrict__ B, OutT* __restrict__ C,
    int M, int N, int K)
{
    __shared__ __attribute__((aligned(16))) u16 lA[128 * 32];  // row-major, stride 32
    __shared__ __attribute__((aligned(16))) u16 lB[256 * 32];  // transposed: lB[n*32+k]

    const int t    = threadIdx.x;
    const int wave = t >> 5;
    const int lane = t & 31;
    const int half = lane >> 4;
    const int l15  = lane & 15;
    const int wm   = wave >> 2;  // 0..1 -> 64 rows each
    const int wn   = wave & 3;   // 0..3 -> 64 cols each
    const int rowBase = blockIdx.y * 128;
    const int colBase = blockIdx.x * 256;

    v8f acc[4][4];
#pragma unroll
    for (int i = 0; i < 4; i++)
#pragma unroll
        for (int j = 0; j < 4; j++) acc[i][j] = splat8(0.f);

    // per-thread staging coordinates
    const int ar = t >> 1;            // A row 0..127
    const int ac = (t & 1) * 16;      // A col 0 / 16
    const int bk = t >> 3;            // B row (k) 0..31
    const int bn = (t & 7) * 32;      // B col base 0..224

    const unsigned ldsA = lds_off_of(lA);

    for (int k0 = 0; k0 < K; k0 += 32) {
        // ---- stage A tile 128x32 via async copies (16 bf16 / thread) ----
        {
            const u16* src = A + (size_t)(rowBase + ar) * K + k0 + ac;
            unsigned dst = ldsA + (unsigned)(ar * 32 + ac) * 2;
            async_copy_b128(src,     dst);
            async_copy_b128(src + 8, dst + 16);
        }
        // ---- stage B tile 32x256 transposed (32 bf16 / thread) ----
        {
            const u16* src = B + (size_t)(k0 + bk) * N + colBase + bn;
#pragma unroll
            for (int j = 0; j < 4; j++) {
                __attribute__((aligned(16))) u16 tmp[8];
                *(U128*)tmp = *(const U128*)(src + j * 8);
#pragma unroll
                for (int i = 0; i < 8; i++) lB[(bn + j * 8 + i) * 32 + bk] = tmp[i];
            }
        }
        if (k0 + 32 < K) {  // prefetch next K-slab
            __builtin_prefetch(A + (size_t)(rowBase + ar) * K + k0 + 32 + ac, 0, 3);
            __builtin_prefetch(B + (size_t)(k0 + 32 + bk) * N + colBase + bn, 0, 3);
        }
        wait_asynccnt0();
        __syncthreads();

        v16bf af[4], bfr[4];
#pragma unroll
        for (int mt = 0; mt < 4; mt++) {
            int r = wm * 64 + mt * 16 + l15;
            FragU f;
            f.h[0] = *(const U128*)(lA + r * 32 + half * 8);        // K 0-7 / 8-15
            f.h[1] = *(const U128*)(lA + r * 32 + 16 + half * 8);   // K 16-23 / 24-31
            af[mt] = f.v;
        }
#pragma unroll
        for (int nt = 0; nt < 4; nt++) {
            int c = wn * 64 + nt * 16 + l15;
            FragU f;
            f.h[0] = *(const U128*)(lB + c * 32 + half * 8);        // K 0-7 / 8-15
            f.h[1] = *(const U128*)(lB + c * 32 + 16 + half * 8);   // K 16-23 / 24-31
            bfr[nt] = f.v;
        }
#pragma unroll
        for (int mt = 0; mt < 4; mt++)
#pragma unroll
            for (int nt = 0; nt < 4; nt++)
                acc[mt][nt] = wmma_bf16(af[mt], bfr[nt], acc[mt][nt]);
        __syncthreads();
    }

    // epilogue: C/D layout -> global
#pragma unroll
    for (int mt = 0; mt < 4; mt++)
#pragma unroll
        for (int nt = 0; nt < 4; nt++)
#pragma unroll
            for (int v = 0; v < 8; v++) {
                int r = rowBase + wm * 64 + mt * 16 + half * 8 + v;
                int c = colBase + wn * 64 + nt * 16 + l15;
                store_out(C + (size_t)r * N + c, acc[mt][nt][v]);
            }
}

// ---------------------------------------------------------------------------
// Flash attention, non-causal. One block = (head, 64 query rows), 4 waves,
// each wave owns a 16-row query tile. K-blocks of 32 keys.
// Q:[S,2048] bf16, K/V:[S,512] bf16, O:[S,2048] bf16.
// ---------------------------------------------------------------------------
__global__ void __launch_bounds__(128) flash_attn_kernel(
    const u16* __restrict__ Q, const u16* __restrict__ Km,
    const u16* __restrict__ Vm, u16* __restrict__ O)
{
    __shared__ __attribute__((aligned(16))) u16 lVT[64 * 32];     // lVT[dcol*32+k]
    __shared__ __attribute__((aligned(16))) u16 lP[4][16 * 32];   // per-wave P tile

    const int h    = blockIdx.y;
    const int g    = h >> 2;             // HEADS_PER_GROUP = 4
    const int t    = threadIdx.x;
    const int wave = t >> 5;
    const int lane = t & 31;
    const int half = lane >> 4;
    const int l15  = lane & 15;
    const int qTileRow = blockIdx.x * 64 + wave * 16;

    // Q fragments for this wave's 16 rows (d=64 -> 2 K-steps of 32)
    v16bf qf[2];
#pragma unroll
    for (int kt = 0; kt < 2; kt++) {
        const u16* qp = Q + (size_t)(qTileRow + l15) * DMODEL + h * HD + kt * 32;
        FragU f;
        f.h[0] = *(const U128*)(qp + half * 8);
        f.h[1] = *(const U128*)(qp + 16 + half * 8);
        qf[kt] = f.v;
    }

    v8f acc[4];
#pragma unroll
    for (int i = 0; i < 4; i++) acc[i] = splat8(0.f);
    v8f mrow = splat8(-1e30f);
    v8f lrow = splat8(0.f);

    for (int kb = 0; kb < S_LEN; kb += 32) {
        // stage V tile (32 keys x 64 d) transposed into LDS
        {
            int k  = t >> 2;          // 0..31
            int nb = (t & 3) * 16;    // 0,16,32,48
            const u16* src = Vm + (size_t)(kb + k) * DKV + g * HD + nb;
            __attribute__((aligned(16))) u16 tmp[16];
            *(U128*)(tmp)     = *(const U128*)(src);
            *(U128*)(tmp + 8) = *(const U128*)(src + 8);
#pragma unroll
            for (int i = 0; i < 16; i++) lVT[(nb + i) * 32 + k] = tmp[i];
        }
        __syncthreads();

        // scores: S = Q * K^T * scale, two 16-key tiles
        v8f sc[2];
#pragma unroll
        for (int nt = 0; nt < 2; nt++) {
            int krow = kb + nt * 16 + l15;
            const u16* kp = Km + (size_t)krow * DKV + g * HD;
            v8f st = splat8(0.f);
#pragma unroll
            for (int kt = 0; kt < 2; kt++) {
                FragU f;
                f.h[0] = *(const U128*)(kp + kt * 32 + half * 8);
                f.h[1] = *(const U128*)(kp + kt * 32 + 16 + half * 8);
                st = wmma_bf16(qf[kt], f.v, st);
            }
            sc[nt] = st * 0.125f;  // 1/sqrt(64)
        }

        // online softmax bookkeeping (rows live per-lane-half, 8 per VGPR)
        v8f rmax = vmax8(sc[0], sc[1]);
#pragma unroll
        for (int mask = 1; mask <= 8; mask <<= 1)
#pragma unroll
            for (int i = 0; i < 8; i++)
                rmax[i] = fmaxf(rmax[i], __shfl_xor(rmax[i], mask, 32));
        v8f mnew = vmax8(mrow, rmax);
        v8f alpha, p0, p1;
#pragma unroll
        for (int i = 0; i < 8; i++) {
            alpha[i] = __expf(mrow[i] - mnew[i]);
            p0[i] = __expf(sc[0][i] - mnew[i]);
            p1[i] = __expf(sc[1][i] - mnew[i]);
        }
        v8f rsum = p0 + p1;
#pragma unroll
        for (int mask = 1; mask <= 8; mask <<= 1)
#pragma unroll
            for (int i = 0; i < 8; i++)
                rsum[i] += __shfl_xor(rsum[i], mask, 32);
#pragma unroll
        for (int i = 0; i < 8; i++) lrow[i] = lrow[i] * alpha[i] + rsum[i];
        mrow = mnew;
#pragma unroll
        for (int j = 0; j < 4; j++) acc[j] = acc[j] * alpha;

        // P (C/D layout f32) -> bf16 A-layout via per-wave LDS round-trip
        u16* pw = lP[wave];
#pragma unroll
        for (int v = 0; v < 8; v++) {
            int r = half * 8 + v;
            pw[r * 32 + l15]      = f32_to_bf16(p0[v]);
            pw[r * 32 + 16 + l15] = f32_to_bf16(p1[v]);
        }
        FragU pf;
        pf.h[0] = *(const U128*)(pw + l15 * 32 + half * 8);
        pf.h[1] = *(const U128*)(pw + l15 * 32 + 16 + half * 8);

        // O += P * V  (N = d = 64 -> 4 tiles, contract over 32 keys)
#pragma unroll
        for (int nt2 = 0; nt2 < 4; nt2++) {
            int c = nt2 * 16 + l15;
            FragU vf;
            vf.h[0] = *(const U128*)(lVT + c * 32 + half * 8);
            vf.h[1] = *(const U128*)(lVT + c * 32 + 16 + half * 8);
            acc[nt2] = wmma_bf16(pf.v, vf.v, acc[nt2]);
        }
        __syncthreads();
    }

    // normalize and write attention output (bf16)
    v8f rl;
#pragma unroll
    for (int i = 0; i < 8; i++) rl[i] = 1.0f / lrow[i];
#pragma unroll
    for (int nt2 = 0; nt2 < 4; nt2++)
#pragma unroll
        for (int v = 0; v < 8; v++) {
            int r = qTileRow + half * 8 + v;
            int c = h * HD + nt2 * 16 + l15;
            O[(size_t)r * DMODEL + c] = f32_to_bf16(acc[nt2][v] * rl[v]);
        }
}

// ---------------------------------------------------------------------------
// launch
// ---------------------------------------------------------------------------
extern "C" void kernel_launch(void* const* d_in, const int* in_sizes, int n_in,
                              void* d_out, int out_size, void* d_ws, size_t ws_size,
                              hipStream_t stream) {
    const float* x  = (const float*)d_in[0];
    const float* wq = (const float*)d_in[1];
    const float* wk = (const float*)d_in[2];
    const float* wv = (const float*)d_in[3];
    const float* wo = (const float*)d_in[4];
    float* out = (float*)d_out;

    char* ws = (char*)d_ws;
    const size_t MB = 1u << 20;
    u16* xb  = (u16*)(ws + 0 * MB);    // 8 MB
    u16* wqb = (u16*)(ws + 8 * MB);    // 8 MB
    u16* wkb = (u16*)(ws + 16 * MB);   // 2 MB
    u16* wvb = (u16*)(ws + 18 * MB);   // 2 MB
    u16* wob = (u16*)(ws + 20 * MB);   // 8 MB
    u16* Qb  = (u16*)(ws + 28 * MB);   // 8 MB
    u16* Kb  = (u16*)(ws + 36 * MB);   // 2 MB
    u16* Vb  = (u16*)(ws + 38 * MB);   // 2 MB
    u16* Ab  = (u16*)(ws + 40 * MB);   // 8 MB  (total 48 MB)

    const int SD  = S_LEN * DMODEL;    // 4194304
    const int DDk = DMODEL * DKV;      // 1048576

    cvt_bf16_kernel<<<1024, 256, 0, stream>>>(x,  xb,  SD);
    cvt_bf16_kernel<<<1024, 256, 0, stream>>>(wq, wqb, SD);
    cvt_bf16_kernel<<<1024, 256, 0, stream>>>(wk, wkb, DDk);
    cvt_bf16_kernel<<<1024, 256, 0, stream>>>(wv, wvb, DDk);
    cvt_bf16_kernel<<<1024, 256, 0, stream>>>(wo, wob, SD);

    // Q = x @ Wq  (2048x2048x2048)
    gemm_bf16_kernel<u16><<<dim3(DMODEL / 256, S_LEN / 128), 256, 0, stream>>>(
        xb, wqb, Qb, S_LEN, DMODEL, DMODEL);
    // K = x @ Wk  (2048x512x2048)
    gemm_bf16_kernel<u16><<<dim3(DKV / 256, S_LEN / 128), 256, 0, stream>>>(
        xb, wkb, Kb, S_LEN, DKV, DMODEL);
    // V = x @ Wv
    gemm_bf16_kernel<u16><<<dim3(DKV / 256, S_LEN / 128), 256, 0, stream>>>(
        xb, wvb, Vb, S_LEN, DKV, DMODEL);

    // attention: grid (query tiles, heads)
    flash_attn_kernel<<<dim3(S_LEN / 64, NHEADS), 128, 0, stream>>>(Qb, Kb, Vb, Ab);

    // out = attn_out @ Wo  (f32 output)
    gemm_bf16_kernel<float><<<dim3(DMODEL / 256, S_LEN / 128), 256, 0, stream>>>(
        Ab, wob, out, S_LEN, DMODEL, DMODEL);
}
